// RoPEMHSA_90829968375924
// MI455X (gfx1250) — matrix operand order repaired and checked by reading, not compiled, not run on hardware
//
#include <hip/hip_runtime.h>
#include <hip/hip_bf16.h>

// ---------------------------------------------------------------------------
// RoPE multi-head self-attention for MI455X (gfx1250), wave32 + WMMA f16
// + async global->LDS staging (ASYNCcnt) for attention K/V tiles.
//
// WMMA operand layouts used by this code (wave32; A and B share one k order):
//  A (16x32 f16, MxK): lane L holds row M=L%16;
//      v16h elems [0..7]  = K (L/16)*16 .. +7
//      v16h elems [8..15] = K (L/16)*16+8 .. +7
//  B (32x16 f16, KxN): lane L holds col N=L%16;
//      v16h elems [0..15] = K (L/16)*16 .. +15   (contiguous per lane)
//  C/D (16x16 f32): lane L holds col N=L%16; VGPR v holds row M = v + 8*(L/16)
// ---------------------------------------------------------------------------

typedef __attribute__((ext_vector_type(16))) _Float16 v16h;
typedef __attribute__((ext_vector_type(8)))  _Float16 v8h;
typedef __attribute__((ext_vector_type(8)))  float    v8f;

#define B_   16
#define S_   1024
#define E_   768
#define H_   12
#define D_   64
#define N3_  2304   // 3*E
#define PI_F 3.14159265358979323846f

__device__ __forceinline__ v8f wmma16(v16h a, v16h b, v8f c) {
  // (neg_a, A, neg_b, B, c_mod, C, reuse_a, reuse_b)
  return __builtin_amdgcn_wmma_f32_16x16x32_f16(false, a, false, b, (short)0, c,
                                                false, false);
}

// A-fragment from row-major f16 [M][ld], rows m_base..+15, cols k_base..+31
__device__ __forceinline__ v16h load_a_frag(const _Float16* p, int ld,
                                            int m_base, int k_base, int lane) {
  const int hs = lane >> 4;
  const _Float16* row = p + (size_t)(m_base + (lane & 15)) * ld + k_base + hs * 16;
  v8h lo = *(const v8h*)(row);        // K = hs*16 .. +7
  v8h hi = *(const v8h*)(row + 8);    // K = hs*16 + 8 .. +15
  v16h out;
#pragma unroll
  for (int i = 0; i < 8; ++i) { out[i] = lo[i]; out[8 + i] = hi[i]; }
  return out;
}

// B-fragment from "N-major" f16 storage Bt[n][k] (ld in halfs):
// lane L -> N = n_base + L%16, K = k_base + (L/16)*16 .. +15 (contiguous)
__device__ __forceinline__ v16h load_b_frag(const _Float16* p, int ld,
                                            int n_base, int k_base, int lane) {
  const _Float16* row =
      p + (size_t)(n_base + (lane & 15)) * ld + k_base + (lane >> 4) * 16;
  return *(const v16h*)(row);
}

// 16-lane-contiguous fragment (16 halfs) from LDS, 16B-aligned pieces
__device__ __forceinline__ v16h ld_lds_frag16(const _Float16* p) {
  v8h lo = *(const v8h*)(p);
  v8h hi = *(const v8h*)(p + 8);
  v16h o;
#pragma unroll
  for (int i = 0; i < 8; ++i) { o[i] = lo[i]; o[8 + i] = hi[i]; }
  return o;
}

// Byte offset of a __shared__ object within the wave's LDS allocation.
// (AS(3) pointers on AMDGPU are 32-bit LDS offsets.)
__device__ __forceinline__ unsigned lds_off(const void* p) {
  return (unsigned)(size_t)(__attribute__((address_space(3))) const void*)p;
}

// Async DMA: 16 bytes global -> LDS, tracked with ASYNCcnt.
__device__ __forceinline__ void async_cp16(unsigned lds_byte_off, const void* g) {
  asm volatile("global_load_async_to_lds_b128 %0, %1, off"
               :: "v"(lds_byte_off), "v"((unsigned long long)(size_t)g)
               : "memory");
}

__device__ __forceinline__ void wait_async0() {
  asm volatile("s_wait_asynccnt 0x0" ::: "memory");
}

// ------------------------------- prep kernels ------------------------------

__global__ void cvt_f16_kernel(const float* __restrict__ src,
                               _Float16* __restrict__ dst, int n) {
  int i = blockIdx.x * 256 + threadIdx.x;
  if (i < n) dst[i] = (_Float16)src[i];
}

// w[K][N] f32 -> wT[N][K] f16
__global__ void transpose_cvt_kernel(const float* __restrict__ w,
                                     _Float16* __restrict__ wT, int K, int N) {
  int i = blockIdx.x * 256 + threadIdx.x;
  if (i < K * N) {
    int n = i / K, k = i % K;
    wT[i] = (_Float16)w[(size_t)k * N + n];
  }
}

// --------------------------- QKV GEMM + RoPE -------------------------------
// grid (N3/128, (B*S)/128), 256 threads (8 waves), wave tile 32(M) x 64(N)
__global__ __launch_bounds__(256)
void qkv_gemm_rope(const _Float16* __restrict__ xh, const _Float16* __restrict__ wT,
                   _Float16* __restrict__ Qb, _Float16* __restrict__ Kb,
                   _Float16* __restrict__ Vt) {
  const int lane = threadIdx.x & 31, wave = threadIdx.x >> 5;
  const int hs = lane >> 4, ln = lane & 15;
  const int m_blk = blockIdx.y * 128, n_blk = blockIdx.x * 128;
  const int wm = m_blk + (wave >> 1) * 32;   // 4 waves along M
  const int wn = n_blk + (wave & 1) * 64;    // 2 waves along N

  v8f acc[2][4];
#pragma unroll
  for (int i = 0; i < 2; ++i)
#pragma unroll
    for (int j = 0; j < 4; ++j)
#pragma unroll
      for (int v = 0; v < 8; ++v) acc[i][j][v] = 0.0f;

  for (int kb = 0; kb < E_; kb += 32) {
    __builtin_prefetch(xh + (size_t)(wm + ln) * E_ + kb + 32, 0, 3);
    v16h a0 = load_a_frag(xh, E_, wm,      kb, lane);
    v16h a1 = load_a_frag(xh, E_, wm + 16, kb, lane);
    v16h b[4];
#pragma unroll
    for (int j = 0; j < 4; ++j) b[j] = load_b_frag(wT, E_, wn + j * 16, kb, lane);
#pragma unroll
    for (int j = 0; j < 4; ++j) {
      acc[0][j] = wmma16(a0, b[j], acc[0][j]);
      acc[1][j] = wmma16(a1, b[j], acc[1][j]);
    }
  }

  // Epilogue: route column c = t*768 + h*64 + d; RoPE on q,k for d<32.
#pragma unroll
  for (int i = 0; i < 2; ++i) {
#pragma unroll
    for (int j = 0; j < 4; ++j) {
      const int gn = wn + j * 16 + ln;
      const int t = gn / E_;
      const int r = gn % E_;
      const int h = r >> 6;
      const int d = r & 63;
#pragma unroll
      for (int v = 0; v < 8; ++v) {
        const int gm = wm + i * 16 + v + 8 * hs;     // global token row
        const int bidx = gm >> 10;
        const int s = gm & 1023;
        float val = acc[i][j][v];
        // partner column d^1 sits in lane^1 of the same 16-lane group
        float partner = __shfl_xor(val, 1);
        if (t < 2 && d < 32) {
          int jj; float pos;
          if (d < 16) { jj = d >> 1;        pos = -1.0f + (2.0f / 31.0f) * (float)(s >> 5); }
          else        { jj = (d - 16) >> 1; pos = -1.0f + (2.0f / 31.0f) * (float)(s & 31); }
          const float th = pos * PI_F * (float)(jj + 1);
          const float cs = __cosf(th), sn = __sinf(th);
          val = val * cs + ((d & 1) ? partner : -partner) * sn;
        }
        const _Float16 hv = (_Float16)val;
        const size_t bh = (size_t)bidx * H_ + h;
        if (t == 0)      Qb[(bh * S_ + s) * D_ + d] = hv;
        else if (t == 1) Kb[(bh * S_ + s) * D_ + d] = hv;
        else             Vt[(bh * D_ + d) * S_ + s] = hv;  // V pre-transposed
      }
    }
  }
}

// ----------------------------- attention -----------------------------------
// grid (S/128, H, B), 256 threads; wave w owns 16 query rows, streams 32-key
// tiles. K/V tiles staged in LDS via double-buffered async global->LDS DMA
// (copy of tile j+1 overlaps compute of tile j). Score->P fragment conversion
// bounces through wave-private LDS. Online softmax by lanes 0..15.
__global__ __launch_bounds__(256)
void attn_kernel(const _Float16* __restrict__ Qb, const _Float16* __restrict__ Kb,
                 const _Float16* __restrict__ Vt, _Float16* __restrict__ Oa) {
  __shared__ _Float16 Ksh[2][32][72];  // [buf][key][dim], 144B rows (bank spread)
  __shared__ _Float16 Vsh[2][64][40];  // [buf][dim][key], 80B rows (bank spread)
  __shared__ float    sS[8][16][36];   // raw scores, wave-private
  __shared__ _Float16 pS[8][16][40];   // exp(p) f16, wave-private
  __shared__ float    fS[8][16];       // per-row rescale / 1/l broadcast

  const int tid = threadIdx.x;
  const int lane = tid & 31, wave = tid >> 5;
  const int hs = lane >> 4, ln = lane & 15;
  const int b = blockIdx.z, h = blockIdx.y;
  const size_t bh = (size_t)b * H_ + h;
  const int q0 = blockIdx.x * 128 + wave * 16;

  const _Float16* Qp = Qb + bh * S_ * D_;
  const _Float16* Kp = Kb + bh * S_ * D_;
  const _Float16* Vp = Vt + bh * D_ * S_;

  const unsigned kshBase = lds_off(&Ksh[0][0][0]);
  const unsigned vshBase = lds_off(&Vsh[0][0][0]);
  const int krow = tid >> 3, kc = tid & 7;   // K tile: 32 rows x 8 16B-chunks
  const int vrow = tid >> 2, vc = tid & 3;   // V tile: 64 rows x 4 16B-chunks

  // issue async copies for tile 0
  async_cp16(kshBase + (unsigned)(krow * 72 * 2 + kc * 16),
             Kp + (size_t)krow * D_ + kc * 8);
  async_cp16(vshBase + (unsigned)(vrow * 40 * 2 + vc * 16),
             Vp + (size_t)vrow * S_ + vc * 8);

  const v16h aq0 = load_a_frag(Qp, D_, q0, 0,  lane);
  const v16h aq1 = load_a_frag(Qp, D_, q0, 32, lane);

  v8f of[4];
#pragma unroll
  for (int t = 0; t < 4; ++t)
#pragma unroll
    for (int v = 0; v < 8; ++v) of[t][v] = 0.0f;
  float m_run = -1e30f, l_run = 0.0f;

  for (int j = 0; j < S_ / 32; ++j) {
    const int k0 = j * 32;
    const int buf = j & 1;

    wait_async0();        // this wave's copies for tile j complete
    __syncthreads();      // all waves' copies visible

    if (j + 1 < S_ / 32) {  // prefetch tile j+1 into the other buffer (overlaps compute)
      const int nbuf = buf ^ 1, nk0 = k0 + 32;
      async_cp16(kshBase + (unsigned)((nbuf * 32 + krow) * 72 * 2 + kc * 16),
                 Kp + (size_t)(nk0 + krow) * D_ + kc * 8);
      async_cp16(vshBase + (unsigned)((nbuf * 64 + vrow) * 40 * 2 + vc * 16),
                 Vp + (size_t)vrow * S_ + nk0 + vc * 8);
    }

    // scores: S(16x32) = Q(16x64) @ K_tile^T, from LDS
    v8f sc0, sc1;
#pragma unroll
    for (int v = 0; v < 8; ++v) { sc0[v] = 0.0f; sc1[v] = 0.0f; }
    sc0 = wmma16(aq0, ld_lds_frag16(&Ksh[buf][ln][hs * 16]),           sc0);
    sc0 = wmma16(aq1, ld_lds_frag16(&Ksh[buf][ln][32 + hs * 16]),      sc0);
    sc1 = wmma16(aq0, ld_lds_frag16(&Ksh[buf][16 + ln][hs * 16]),      sc1);
    sc1 = wmma16(aq1, ld_lds_frag16(&Ksh[buf][16 + ln][32 + hs * 16]), sc1);

#pragma unroll
    for (int v = 0; v < 8; ++v) {
      const int r = v + 8 * hs;
      sS[wave][r][ln]      = sc0[v];
      sS[wave][r][16 + ln] = sc1[v];
    }
    __syncthreads();

    if (lane < 16) {   // lane r handles query row r of this wave
      const int r = lane;
      float mx = m_run;
#pragma unroll
      for (int c = 0; c < 32; ++c) mx = fmaxf(mx, sS[wave][r][c] * 0.125f);
      const float alpha = __expf(m_run - mx);
      float ssum = 0.0f;
#pragma unroll
      for (int c = 0; c < 32; ++c) {
        const float p = __expf(sS[wave][r][c] * 0.125f - mx);
        ssum += p;
        pS[wave][r][c] = (_Float16)p;
      }
      l_run = l_run * alpha + ssum;
      m_run = mx;
      fS[wave][r] = alpha;
    }
    __syncthreads();

    // rescale O accumulator, then O += P @ V (V from LDS)
    float fv[8];
#pragma unroll
    for (int v = 0; v < 8; ++v) fv[v] = fS[wave][v + 8 * hs];
#pragma unroll
    for (int t = 0; t < 4; ++t)
#pragma unroll
      for (int v = 0; v < 8; ++v) of[t][v] = of[t][v] * fv[v];

    // P as A-fragment from wave-private LDS
    const _Float16* prow = &pS[wave][ln][hs * 16];
    v8h lo = *(const v8h*)(prow);
    v8h hi = *(const v8h*)(prow + 8);
    v16h pa;
#pragma unroll
    for (int i = 0; i < 8; ++i) { pa[i] = lo[i]; pa[8 + i] = hi[i]; }

#pragma unroll
    for (int t = 0; t < 4; ++t)
      of[t] = wmma16(pa, ld_lds_frag16(&Vsh[buf][t * 16 + ln][hs * 16]), of[t]);
    __syncthreads();   // done reading buf before tile j+2's DMA overwrites it
  }

  if (lane < 16) fS[wave][lane] = 1.0f / l_run;
  __syncthreads();

#pragma unroll
  for (int v = 0; v < 8; ++v) {
    const float inv = fS[wave][v + 8 * hs];
    const int tok = q0 + v + 8 * hs;
#pragma unroll
    for (int t = 0; t < 4; ++t)
      Oa[((size_t)b * S_ + tok) * E_ + h * D_ + t * 16 + ln] =
          (_Float16)(of[t][v] * inv);
  }
}

// --------------------------- output projection -----------------------------
// grid (E/128, (B*S)/128), 256 threads; out = Oa @ w_proj + b_proj (f32 out)
__global__ __launch_bounds__(256)
void proj_gemm(const _Float16* __restrict__ Oa, const _Float16* __restrict__ wT,
               const float* __restrict__ bias, float* __restrict__ out) {
  const int lane = threadIdx.x & 31, wave = threadIdx.x >> 5;
  const int hs = lane >> 4, ln = lane & 15;
  const int m_blk = blockIdx.y * 128, n_blk = blockIdx.x * 128;
  const int wm = m_blk + (wave >> 1) * 32;
  const int wn = n_blk + (wave & 1) * 64;

  v8f acc[2][4];
#pragma unroll
  for (int i = 0; i < 2; ++i)
#pragma unroll
    for (int j = 0; j < 4; ++j)
#pragma unroll
      for (int v = 0; v < 8; ++v) acc[i][j][v] = 0.0f;

  for (int kb = 0; kb < E_; kb += 32) {
    __builtin_prefetch(Oa + (size_t)(wm + ln) * E_ + kb + 32, 0, 3);
    v16h a0 = load_a_frag(Oa, E_, wm,      kb, lane);
    v16h a1 = load_a_frag(Oa, E_, wm + 16, kb, lane);
    v16h bfr[4];
#pragma unroll
    for (int j = 0; j < 4; ++j) bfr[j] = load_b_frag(wT, E_, wn + j * 16, kb, lane);
#pragma unroll
    for (int j = 0; j < 4; ++j) {
      acc[0][j] = wmma16(a0, bfr[j], acc[0][j]);
      acc[1][j] = wmma16(a1, bfr[j], acc[1][j]);
    }
  }

#pragma unroll
  for (int j = 0; j < 4; ++j) {
    const int gn = wn + j * 16 + ln;
    const float bv = bias[gn];
#pragma unroll
    for (int i = 0; i < 2; ++i)
#pragma unroll
      for (int v = 0; v < 8; ++v) {
        const int gm = wm + i * 16 + v + 8 * hs;
        out[(size_t)gm * E_ + gn] = acc[i][j][v] + bv;
      }
  }
}

// ------------------------------- launcher ----------------------------------

extern "C" void kernel_launch(void* const* d_in, const int* in_sizes, int n_in,
                              void* d_out, int out_size, void* d_ws, size_t ws_size,
                              hipStream_t stream) {
  const float* x      = (const float*)d_in[0];
  const float* w_qkv  = (const float*)d_in[1];
  const float* w_proj = (const float*)d_in[2];
  const float* b_proj = (const float*)d_in[3];
  float* out = (float*)d_out;

  char* ws = (char*)d_ws;
  // workspace layout (bytes); all offsets 256B-aligned
  _Float16* xh     = (_Float16*)(ws);                 // 16384*768*2   = 25,165,824
  _Float16* wqkvT  = (_Float16*)(ws + 25165824);      // 2304*768*2    =  3,538,944
  _Float16* wprojT = (_Float16*)(ws + 28704768);      // 768*768*2     =  1,179,648
  _Float16* Qb     = (_Float16*)(ws + 29884416);      // B*H*S*D*2     = 25,165,824
  _Float16* Kb     = (_Float16*)(ws + 55050240);      // 25,165,824
  _Float16* Vt     = (_Float16*)(ws + 80216064);      // 25,165,824 ([B,H,D,S])
  _Float16* Oa     = (_Float16*)(ws + 105381888);     // 25,165,824
  // total ~130.5 MB (fits the 192MB L2)

  {
    int n = B_ * S_ * E_;
    cvt_f16_kernel<<<(n + 255) / 256, 256, 0, stream>>>(x, xh, n);
  }
  {
    int n = E_ * N3_;
    transpose_cvt_kernel<<<(n + 255) / 256, 256, 0, stream>>>(w_qkv, wqkvT, E_, N3_);
  }
  {
    int n = E_ * E_;
    transpose_cvt_kernel<<<(n + 255) / 256, 256, 0, stream>>>(w_proj, wprojT, E_, E_);
  }
  qkv_gemm_rope<<<dim3(N3_ / 128, (B_ * S_) / 128), 256, 0, stream>>>(
      xh, wqkvT, Qb, Kb, Vt);
  attn_kernel<<<dim3(S_ / 128, H_, B_), 256, 0, stream>>>(Qb, Kb, Vt, Oa);
  proj_gemm<<<dim3(E_ / 128, (B_ * S_) / 128), 256, 0, stream>>>(
      Oa, wprojT, b_proj, out);
}
